// SelfAttentionBlock_83176336654650
// MI455X (gfx1250) — matrix operand order repaired and checked by reading, not compiled, not run on hardware
//
#include <hip/hip_runtime.h>

typedef _Float16 half_t;
typedef __attribute__((ext_vector_type(16))) _Float16 v16h;
typedef __attribute__((ext_vector_type(8)))  float    v8f;
typedef __attribute__((ext_vector_type(4)))  unsigned int u32x4;
typedef __attribute__((ext_vector_type(8)))  int          i32x8;
typedef __attribute__((ext_vector_type(4)))  int          i32x4;

constexpr int BATCH = 4;
constexpr int CH    = 256;   // C
constexpr int RCH   = 128;   // RC (head dim)
constexpr int NN    = 4096;  // H*W

union frag_u { v16h v; uint4 q[2]; };

static __device__ __forceinline__ v8f wmma_f16(v16h a, v16h b, v8f c) {
  return __builtin_amdgcn_wmma_f32_16x16x32_f16(false, a, false, b, (short)0, c,
                                                false, false);
}

// Async global->LDS 128-bit copy (GLOBAL_LOAD_ASYNC_TO_LDS_B128, ASYNCcnt).
static __device__ __forceinline__ void async_g2l_b128(half_t* lds,
                                                      const half_t* gsrc) {
  unsigned laddr = (unsigned)(unsigned long long)lds;
  unsigned long long gaddr = (unsigned long long)gsrc;
  asm volatile("global_load_async_to_lds_b128 %0, %1, off"
               :: "v"(laddr), "v"(gaddr) : "memory");
}
static __device__ __forceinline__ void wait_async0() {
  asm volatile("s_wait_asynccnt 0x0" ::: "memory");
}
static __device__ __forceinline__ void wait_async_le4() {
  asm volatile("s_wait_asynccnt 0x4" ::: "memory");
}

#if __has_builtin(__builtin_amdgcn_tensor_load_to_lds)
#define USE_TDM 1
// Build a 2D Tensor-DMA descriptor (ISA ch.8) and issue TENSOR_LOAD_TO_LDS.
//   tile (tile_d1 rows x tile_d0 elems, f16) from row-major tensor with
//   dim0 stride `stride0` (elems) into contiguous LDS at `lds`.
// This toolchain declares the 6-arg builtin:
//   (u32x4 g0, i32x8 g1, i32x4 g2, i32x4 g3, i32x8 g23hi, i32 cpol)
static __device__ __forceinline__ void tdm_load_2d(
    void* lds, const half_t* gsrc,
    unsigned tensor_d0, unsigned tensor_d1, unsigned long long stride0,
    unsigned tile_d0, unsigned tile_d1)
{
  unsigned long long ga = (unsigned long long)gsrc;
  u32x4 g0;
  g0[0] = 1u;                                        // count=1, user mode
  g0[1] = (unsigned)(unsigned long long)lds;         // lds_addr
  g0[2] = (unsigned)(ga & 0xffffffffu);              // global_addr[31:0]
  g0[3] = (unsigned)((ga >> 32) & 0x01ffffffu) | (2u << 30);  // [56:32]|type=2
  i32x8 g1;
  g1[0] = (int)(1u << 16);                           // data_size=1 (2 bytes)
  g1[1] = (int)((tensor_d0 & 0xffffu) << 16);        // tensor_dim0[15:0]
  g1[2] = (int)((tensor_d0 >> 16) | ((tensor_d1 & 0xffffu) << 16));
  g1[3] = (int)((tensor_d1 >> 16) | (tile_d0 << 16));// tensor_dim1 hi | tile0
  g1[4] = (int)(tile_d1 & 0xffffu);                  // tile_dim1 (tile_dim2=0)
  g1[5] = (int)(unsigned)(stride0 & 0xffffffffu);    // dim0 stride lo
  g1[6] = (int)(unsigned)((stride0 >> 32) & 0xffffu);// dim0 stride hi
  g1[7] = 0;
  i32x4 z4 = {0, 0, 0, 0};                           // groups 2/3 unused (2D)
  i32x8 z8 = {0, 0, 0, 0, 0, 0, 0, 0};
  __builtin_amdgcn_tensor_load_to_lds(g0, g1, z4, z4, z8, 0);
}
#else
#define USE_TDM 0
#endif

// ---------------------------------------------------------------------------
// Kernel 0: one-shot f32 -> f16 weight conversion (L2-resident thereafter).
// ---------------------------------------------------------------------------
__global__ __launch_bounds__(256) void cvt_weights_kernel(
    const float* __restrict__ wq, const float* __restrict__ wk,
    const float* __restrict__ wv, const float* __restrict__ wo,
    half_t* __restrict__ Wq, half_t* __restrict__ Wk,
    half_t* __restrict__ Wv, half_t* __restrict__ Wo)
{
  const int i = blockIdx.x * 256 + threadIdx.x;   // 32768 threads
  Wq[i] = (half_t)wq[i];
  Wk[i] = (half_t)wk[i];
  Wv[i] = (half_t)wv[i];
  Wo[i] = (half_t)wo[i];
}

// ---------------------------------------------------------------------------
// Kernel 1: fused QKV projections -> Q[b][n][r], K[b][n][r], Vt[b][r][n].
// ---------------------------------------------------------------------------
__global__ __launch_bounds__(256) void qkv_proj_kernel(
    const float* __restrict__ x,
    const half_t* __restrict__ Wq, const float* __restrict__ bq,
    const half_t* __restrict__ Wk, const float* __restrict__ bk,
    const half_t* __restrict__ Wv, const float* __restrict__ bv,
    half_t* __restrict__ Qh, half_t* __restrict__ Kh, half_t* __restrict__ Vt)
{
  __shared__ __align__(32) half_t Xs[64][CH];  // X^T tile [n][c], 32 KB

  const int b   = blockIdx.x;
  const int n0  = blockIdx.y * 64;
  const int tid = threadIdx.x;
  const int lane = tid & 31;
  const int wid  = tid >> 5;
  const int l16  = lane & 15;
  const int hl   = lane >> 4;

  {  // stage X^T tile (f32 -> f16 conversion must go through VGPRs)
    const float* xb = x + ((size_t)b * CH) * NN + n0;
    for (int c = wid; c < CH; c += 8) {
      float a0 = xb[(size_t)c * NN + lane];
      float a1 = xb[(size_t)c * NN + lane + 32];
      Xs[lane][c]      = (half_t)a0;
      Xs[lane + 32][c] = (half_t)a1;
    }
  }
  __syncthreads();

  for (int t = wid; t < 96; t += 8) {
    const int mat = t >> 5;
    const int rt  = (t & 31) >> 2;
    const int nt  = t & 3;
    const half_t* W  = (mat == 0) ? Wq : (mat == 1) ? Wk : Wv;
    const float*  bb = (mat == 0) ? bq : (mat == 1) ? bk : bv;

    v8f acc = {};
    #pragma unroll
    for (int cc = 0; cc < CH; cc += 32) {
      const half_t* wrow = W + (size_t)(rt * 16 + l16) * CH + cc + hl * 16;
      frag_u a;
      a.q[0] = *(const uint4*)(wrow);
      a.q[1] = *(const uint4*)(wrow + 8);
      v16h bf = *(const v16h*)&Xs[nt * 16 + l16][cc + hl * 16];
      acc = wmma_f16(a.v, bf, acc);
    }

    const int rbase = rt * 16 + 8 * hl;
    const int ncol  = n0 + nt * 16 + l16;
    if (mat < 2) {
      half_t* dst = ((mat == 0) ? Qh : Kh) + ((size_t)b * NN + ncol) * RCH + rbase;
      union { half_t h[8]; uint4 u; } pk;
      #pragma unroll
      for (int i = 0; i < 8; ++i) pk.h[i] = (half_t)(acc[i] + bb[rbase + i]);
      *reinterpret_cast<uint4*>(dst) = pk.u;
    } else {
      #pragma unroll
      for (int i = 0; i < 8; ++i)
        Vt[((size_t)b * RCH + rbase + i) * NN + ncol] = (half_t)(acc[i] + bb[rbase + i]);
    }
  }
}

// ---------------------------------------------------------------------------
// Kernel 2: flash attention with double-buffered K/V tiles fed by the Tensor
// Data Mover (TENSORcnt pipelining); online softmax; O in 64 VGPRs.
// ---------------------------------------------------------------------------
__global__ __launch_bounds__(256) void flash_attn_kernel(
    const half_t* __restrict__ Qh, const half_t* __restrict__ Kh,
    const half_t* __restrict__ Vt, half_t* __restrict__ Oh)
{
  __shared__ __align__(32) half_t Ks [2][32][RCH];   // K blocks [m][r], 16 KB
  __shared__ __align__(32) half_t Vls[2][RCH][32];   // V blocks [r][m], 16 KB
  __shared__ __align__(32) half_t Pb [8][16][32];    // per-wave P tile, 8 KB

  const int b    = blockIdx.x;
  const int q0   = blockIdx.y * 128;
  const int tid  = threadIdx.x;
  const int lane = tid & 31;
  const int wid  = tid >> 5;
  const int l16  = lane & 15;
  const int hl   = lane >> 4;

  const int nrow = q0 + wid * 16;

  // Resident Q fragments, with 1/sqrt(RC) folded in once.
  v16h qf[4];
  {
    const half_t hs = (half_t)0.08838834764831845f;
    const half_t* qbase = Qh + ((size_t)b * NN + nrow + l16) * RCH;
    #pragma unroll
    for (int rc = 0; rc < 4; ++rc) {
      const half_t* p = qbase + rc * 32 + hl * 16;
      frag_u f;
      f.q[0] = *(const uint4*)(p);
      f.q[1] = *(const uint4*)(p + 8);
      #pragma unroll
      for (int j = 0; j < 16; ++j) f.v[j] = f.v[j] * hs;
      qf[rc] = f.v;
    }
  }

  float mrow[8], lrow[8];
  v8f o[8] = {};
  #pragma unroll
  for (int i = 0; i < 8; ++i) { mrow[i] = -3.0e38f; lrow[i] = 0.0f; }

  const half_t* kbb = Kh + (size_t)b * NN * RCH;
  const half_t* vbb = Vt + (size_t)b * RCH * NN;
  const int nblk = NN / 32;   // 128

#if USE_TDM
  if (wid == 0) {  // prologue: DMA block 0 into buffer 0
    tdm_load_2d(&Ks[0][0][0],  kbb, RCH, NN, RCH, RCH, 32);
    tdm_load_2d(&Vls[0][0][0], vbb, NN, RCH, NN,  32,  RCH);
  }
#else
  {
    const int m = tid >> 3, seg = tid & 7;
    async_g2l_b128(&Ks[0][m][seg * 16],     kbb + (size_t)m * RCH + seg * 16);
    async_g2l_b128(&Ks[0][m][seg * 16 + 8], kbb + (size_t)m * RCH + seg * 16 + 8);
    const int r = tid >> 1, hm = tid & 1;
    async_g2l_b128(&Vls[0][r][hm * 16],     vbb + (size_t)r * NN + hm * 16);
    async_g2l_b128(&Vls[0][r][hm * 16 + 8], vbb + (size_t)r * NN + hm * 16 + 8);
  }
#endif

  for (int it = 0; it < nblk; ++it) {
    const int buf = it & 1;
    const int kb  = it * 32;

    __syncthreads();  // all waves done reading buf^1 from iteration it-1
#if USE_TDM
    if (wid == 0) {
      if (it + 1 < nblk) {  // DMA next block under this block's compute
        tdm_load_2d(&Ks[buf ^ 1][0][0],  kbb + (size_t)(kb + 32) * RCH,
                    RCH, NN, RCH, RCH, 32);
        tdm_load_2d(&Vls[buf ^ 1][0][0], vbb + (kb + 32),
                    NN, RCH, NN, 32, RCH);
        __builtin_amdgcn_s_wait_tensorcnt(2);  // block `it` complete
      } else {
        __builtin_amdgcn_s_wait_tensorcnt(0);
      }
    }
#else
    if (it + 1 < nblk) {
      const int m = tid >> 3, seg = tid & 7;
      const half_t* ksrc = kbb + (size_t)(kb + 32 + m) * RCH + seg * 16;
      async_g2l_b128(&Ks[buf ^ 1][m][seg * 16],     ksrc);
      async_g2l_b128(&Ks[buf ^ 1][m][seg * 16 + 8], ksrc + 8);
      const int r = tid >> 1, hm = tid & 1;
      const half_t* vsrc = vbb + (size_t)r * NN + kb + 32 + hm * 16;
      async_g2l_b128(&Vls[buf ^ 1][r][hm * 16],     vsrc);
      async_g2l_b128(&Vls[buf ^ 1][r][hm * 16 + 8], vsrc + 8);
      wait_async_le4();   // the 4 oldest (block `it`) have landed
    } else {
      wait_async0();
    }
#endif
    __syncthreads();  // buffer `buf` visible to all waves

    // S = Q K^T (scale pre-folded): two 16x16 tiles, K-dim 128.
    v8f s0 = {}, s1 = {};
    #pragma unroll
    for (int rc = 0; rc < 4; ++rc) {
      v16h k0 = *(const v16h*)&Ks[buf][l16]     [rc * 32 + hl * 16];
      v16h k1 = *(const v16h*)&Ks[buf][16 + l16][rc * 32 + hl * 16];
      s0 = wmma_f16(qf[rc], k0, s0);
      s1 = wmma_f16(qf[rc], k1, s1);
    }

    // Online softmax.
    float corr[8];
    #pragma unroll
    for (int i = 0; i < 8; ++i) {
      float r = fmaxf(s0[i], s1[i]);
      r = fmaxf(r, __shfl_xor(r, 1));
      r = fmaxf(r, __shfl_xor(r, 2));
      r = fmaxf(r, __shfl_xor(r, 4));
      r = fmaxf(r, __shfl_xor(r, 8));
      const float mnew = fmaxf(mrow[i], r);
      corr[i] = __expf(mrow[i] - mnew);
      mrow[i] = mnew;
      s0[i] = __expf(s0[i] - mnew);
      s1[i] = __expf(s1[i] - mnew);
      float rs = s0[i] + s1[i];
      rs += __shfl_xor(rs, 1);
      rs += __shfl_xor(rs, 2);
      rs += __shfl_xor(rs, 4);
      rs += __shfl_xor(rs, 8);
      lrow[i] = lrow[i] * corr[i] + rs;
    }

    // D-layout P -> per-wave LDS -> A-layout fragment (DS in-order per wave).
    #pragma unroll
    for (int i = 0; i < 8; ++i) {
      Pb[wid][i + 8 * hl][l16]      = (half_t)s0[i];
      Pb[wid][i + 8 * hl][16 + l16] = (half_t)s1[i];
    }
    frag_u pf;
    {
      const half_t* pp = &Pb[wid][l16][hl * 16];
      pf.q[0] = *(const uint4*)(pp);
      pf.q[1] = *(const uint4*)(pp + 8);
    }

    // O = O*corr + P V.
    #pragma unroll
    for (int ch = 0; ch < 8; ++ch) {
      #pragma unroll
      for (int i = 0; i < 8; ++i) o[ch][i] *= corr[i];
      v16h vf = *(const v16h*)&Vls[buf][ch * 16 + l16][hl * 16];
      o[ch] = wmma_f16(pf.v, vf, o[ch]);
    }
  }

  #pragma unroll
  for (int i = 0; i < 8; ++i) lrow[i] = 1.0f / lrow[i];
  #pragma unroll
  for (int ch = 0; ch < 8; ++ch) {
    #pragma unroll
    for (int i = 0; i < 8; ++i) {
      const int n = nrow + i + 8 * hl;
      Oh[((size_t)b * NN + n) * RCH + ch * 16 + l16] = (half_t)(o[ch][i] * lrow[i]);
    }
  }
}

// ---------------------------------------------------------------------------
// Kernel 3: output projection + gamma-scaled residual (async-LDS staging).
// ---------------------------------------------------------------------------
__global__ __launch_bounds__(256) void out_proj_kernel(
    const half_t* __restrict__ Oh, const half_t* __restrict__ Wo,
    const float* __restrict__ bo, const float* __restrict__ x,
    const float* __restrict__ gamma, float* __restrict__ y)
{
  __shared__ __align__(32) half_t Os[64][RCH];  // attn-out tile [n][r], 16 KB

  const int b    = blockIdx.x;
  const int n0   = blockIdx.y * 64;
  const int c0   = blockIdx.z * 128;
  const int tid  = threadIdx.x;
  const int lane = tid & 31;
  const int wid  = tid >> 5;
  const int l16  = lane & 15;
  const int hl   = lane >> 4;

  {  // async DMA staging of the O tile
    const int n = tid >> 2, seg = tid & 3;
    const half_t* src = Oh + ((size_t)b * NN + n0 + n) * RCH + seg * 32;
    #pragma unroll
    for (int j = 0; j < 4; ++j)
      async_g2l_b128(&Os[n][seg * 32 + j * 8], src + j * 8);
  }
  wait_async0();
  __syncthreads();

  const float g = gamma[0];
  const int crow = c0 + wid * 16;

  v8f acc[4] = {};
  #pragma unroll
  for (int rc = 0; rc < 4; ++rc) {
    const half_t* wrow = Wo + (size_t)(crow + l16) * RCH + rc * 32 + hl * 16;
    frag_u a;
    a.q[0] = *(const uint4*)(wrow);
    a.q[1] = *(const uint4*)(wrow + 8);
    #pragma unroll
    for (int nt = 0; nt < 4; ++nt) {
      v16h bf = *(const v16h*)&Os[nt * 16 + l16][rc * 32 + hl * 16];
      acc[nt] = wmma_f16(a.v, bf, acc[nt]);
    }
  }

  #pragma unroll
  for (int nt = 0; nt < 4; ++nt) {
    #pragma unroll
    for (int i = 0; i < 8; ++i) {
      const int c = crow + 8 * hl + i;
      const size_t idx = ((size_t)b * CH + c) * NN + n0 + nt * 16 + l16;
      y[idx] = g * (acc[nt][i] + bo[c]) + x[idx];
    }
  }
}

// ---------------------------------------------------------------------------
extern "C" void kernel_launch(void* const* d_in, const int* in_sizes, int n_in,
                              void* d_out, int out_size, void* d_ws, size_t ws_size,
                              hipStream_t stream) {
  (void)in_sizes; (void)n_in; (void)out_size; (void)ws_size;

  const float* x     = (const float*)d_in[0];
  const float* wq    = (const float*)d_in[1];
  const float* bq    = (const float*)d_in[2];
  const float* wk    = (const float*)d_in[3];
  const float* bk    = (const float*)d_in[4];
  const float* wv    = (const float*)d_in[5];
  const float* bv    = (const float*)d_in[6];
  const float* wo    = (const float*)d_in[7];
  const float* bo    = (const float*)d_in[8];
  const float* gamma = (const float*)d_in[9];
  float* y = (float*)d_out;

  const size_t per  = (size_t)BATCH * NN * RCH;   // 2 Mi f16 = 4 MB each
  const size_t wsz  = (size_t)RCH * CH;           // 32768 f16 per weight mat
  half_t* Qh = (half_t*)d_ws;
  half_t* Kh = Qh + per;
  half_t* Vt = Kh + per;
  half_t* Oh = Vt + per;
  half_t* Wq = Oh + per;
  half_t* Wk = Wq + wsz;
  half_t* Wv = Wk + wsz;
  half_t* Wo = Wv + wsz;

  dim3 blk(256);
  cvt_weights_kernel<<<dim3((unsigned)(wsz / 256)), blk, 0, stream>>>(
      wq, wk, wv, wo, Wq, Wk, Wv, Wo);
  qkv_proj_kernel<<<dim3(BATCH, NN / 64), blk, 0, stream>>>(
      x, Wq, bq, Wk, bk, Wv, bv, Qh, Kh, Vt);
  flash_attn_kernel<<<dim3(BATCH, NN / 128), blk, 0, stream>>>(Qh, Kh, Vt, Oh);
  out_proj_kernel<<<dim3(BATCH, NN / 64, CH / 128), blk, 0, stream>>>(
      Oh, Wo, bo, x, gamma, y);
}
